// SparseMoELanguageModel_47201690583045
// MI455X (gfx1250) — compile-verified, matrix operand order checked by reading, not run on hardware
//
#include <hip/hip_runtime.h>
#include <hip/hip_bf16.h>
#include <math.h>

// Problem dims (fixed by the reference)
#define B_    2
#define T_    2048
#define D_    1024
#define H_    16
#define L_    2
#define E_    8
#define V_    32000
#define HS_   64
#define NTOK_ (B_*T_)      // 4096
#define CAP_  1024          // B*T*K/E
#define FF_   4096          // 4*D
#define PAD_  2

typedef __attribute__((ext_vector_type(16))) _Float16 v16h;
typedef __attribute__((ext_vector_type(8)))  _Float16 v8h;
typedef __attribute__((ext_vector_type(8)))  float    v8f;
typedef __attribute__((ext_vector_type(4)))  unsigned int u32x4;
typedef __attribute__((ext_vector_type(8)))  int      i32x8;
typedef __attribute__((ext_vector_type(4)))  int      i32x4;

// ---------------------------------------------------------------------------
// TDM: 2D tile load (rows x cols f16 elements) from global to LDS.
// Descriptor layout per CDNA5 ISA ch.8. Issued per-wave; completion tracked
// by TENSORcnt (s_wait_tensorcnt).
// ---------------------------------------------------------------------------
__device__ __forceinline__ void tdm_load_2d(unsigned lds_addr, const void* gptr,
                                            unsigned rows, unsigned cols,
                                            unsigned stride_elems) {
  unsigned long long ga = (unsigned long long)(uintptr_t)gptr;
  u32x4 g0 = {
    1u,                                   // count=1 (valid user descriptor)
    lds_addr,                             // LDS byte address
    (unsigned)(ga & 0xffffffffu),         // global_addr[31:0]
    (unsigned)((ga >> 32) & 0x1ffffffu) | (2u << 30)  // global_addr[56:32] | type=2
  };
  unsigned t0 = stride_elems;             // tensor_dim0 (no OOB on X)
  unsigned t1 = 1u << 20;                 // tensor_dim1 (no OOB on Y)
  unsigned long long st = (unsigned long long)stride_elems;
  i32x8 g1 = {
    (int)(1u << 16),                                       // data_size=1 (2 bytes)
    (int)((t0 & 0xffffu) << 16),                           // tensor_dim0[15:0]
    (int)((t0 >> 16) | ((t1 & 0xffffu) << 16)),            // dim0 hi | dim1 lo
    (int)((t1 >> 16) | ((cols & 0xffffu) << 16)),          // dim1 hi | tile_dim0
    (int)(rows & 0xffffu),                                 // tile_dim1 (tile_dim2=0)
    (int)(st & 0xffffffffu),                               // dim0_stride[31:0]
    (int)((st >> 32) & 0xffffu),                           // dim0_stride[47:32]
    0
  };
  i32x4 gz = {0, 0, 0, 0};
#if defined(__clang_major__) && __clang_major__ >= 23
  i32x8 gz8 = {0, 0, 0, 0, 0, 0, 0, 0};
  __builtin_amdgcn_tensor_load_to_lds(g0, g1, gz, gz, gz8, 0);
#else
  __builtin_amdgcn_tensor_load_to_lds(g0, g1, gz, gz, 0);
#endif
}

// ---------------------------------------------------------------------------
// WMMA fragment helper (CDNA5 16x16x32 f16, wave32). Operand stored [row][K]
// K-contiguous in LDS; per-lane two contiguous 16B chunks -> 2x ds_load_b128.
// ---------------------------------------------------------------------------
__device__ __forceinline__ v16h load_frag_NK(const _Float16* base, int ld, int row0, int k0) {
  int l = threadIdx.x & 31;
  const _Float16* p = base + (row0 + (l & 15)) * ld + k0 + 8 * (l >> 4);
  v8h lo = *(const v8h*)p;
  v8h hi = *(const v8h*)(p + 16);
  return __builtin_shufflevector(lo, hi, 0, 1, 2, 3, 4, 5, 6, 7,
                                 8, 9, 10, 11, 12, 13, 14, 15);
}

__device__ __forceinline__ v8f wmma16(v16h a, v16h b, v8f c) {
  return __builtin_amdgcn_wmma_f32_16x16x32_f16(false, a, false, b, (short)0, c, false, false);
}

// ---------------------------------------------------------------------------
// Elementwise f32 -> f16 convert
// ---------------------------------------------------------------------------
__global__ void convert_kernel(const float* __restrict__ in, _Float16* __restrict__ out,
                               int n) {
  for (int i = blockIdx.x * blockDim.x + threadIdx.x; i < n; i += gridDim.x * blockDim.x)
    out[i] = (_Float16)in[i];
}

// ---------------------------------------------------------------------------
// Transpose + convert: B[K][N] f32 -> Bt[N][K] f16 (32x32 LDS tiles)
// ---------------------------------------------------------------------------
__global__ void transpose_conv_kernel(const float* __restrict__ B, _Float16* __restrict__ Bt,
                                      int K, int N) {
  __shared__ _Float16 t[32][33];
  int k0 = blockIdx.y * 32, n0 = blockIdx.x * 32;
  int tj = threadIdx.x & 31, ti = threadIdx.x >> 5;   // 256 threads: 8 rows/pass
#pragma unroll
  for (int p = 0; p < 4; ++p) {
    int k = ti + p * 8;
    t[k][tj] = (_Float16)B[(size_t)(k0 + k) * N + n0 + tj];
  }
  __syncthreads();
#pragma unroll
  for (int p = 0; p < 4; ++p) {
    int n = ti + p * 8;
    Bt[(size_t)(n0 + n) * K + k0 + tj] = t[tj][n];
  }
}

// ---------------------------------------------------------------------------
// Embedding: x[n,:] = tok_emb[ids[n],:] + pos_emb[n%T,:]
// ---------------------------------------------------------------------------
__global__ void embed_kernel(const int* __restrict__ ids, const float* __restrict__ tok,
                             const float* __restrict__ pos, float* __restrict__ x) {
  int n = blockIdx.x;
  int t = n % T_;
  int id = ids[n];
  for (int d = threadIdx.x; d < D_; d += 256)
    x[(size_t)n * D_ + d] = tok[(size_t)id * D_ + d] + pos[(size_t)t * D_ + d];
}

// ---------------------------------------------------------------------------
// LayerNorm (one block per row); writes f32 (router) and f16 (GEMM/NLL) copies
// ---------------------------------------------------------------------------
__global__ void ln_kernel(const float* __restrict__ x, const float* __restrict__ w,
                          const float* __restrict__ b, float* __restrict__ y,
                          _Float16* __restrict__ yh) {
  int row = blockIdx.x;
  const float* xr = x + (size_t)row * D_;
  __shared__ float red[256];
  float s = 0.f;
  for (int d = threadIdx.x; d < D_; d += 256) s += xr[d];
  red[threadIdx.x] = s; __syncthreads();
  for (int off = 128; off > 0; off >>= 1) {
    if (threadIdx.x < off) red[threadIdx.x] += red[threadIdx.x + off];
    __syncthreads();
  }
  float mu = red[0] / (float)D_;
  __syncthreads();
  float v = 0.f;
  for (int d = threadIdx.x; d < D_; d += 256) { float t = xr[d] - mu; v += t * t; }
  red[threadIdx.x] = v; __syncthreads();
  for (int off = 128; off > 0; off >>= 1) {
    if (threadIdx.x < off) red[threadIdx.x] += red[threadIdx.x + off];
    __syncthreads();
  }
  float inv = rsqrtf(red[0] / (float)D_ + 1e-5f);
  for (int d = threadIdx.x; d < D_; d += 256) {
    float o = (xr[d] - mu) * inv * w[d] + b[d];
    y[(size_t)row * D_ + d] = o;
    yh[(size_t)row * D_ + d] = (_Float16)o;
  }
}

// ---------------------------------------------------------------------------
// WMMA GEMM, double-buffered TDM pipeline:
//   C[M,N] = A[M,K] @ Bt[N,K]^T (+bias) (opt relu)
// Wave 0 issues the next tile's TDM loads, then s_wait_tensorcnt(2) so the
// current tile (issued one iteration earlier, in-order completion) is ready
// while the next stays in flight. M,N multiples of 64; K multiple of 32.
// ---------------------------------------------------------------------------
template <bool BIAS, bool RELU, bool HALF_OUT>
__global__ __launch_bounds__(128) void gemm_tdm(const _Float16* __restrict__ A,
                                                const _Float16* __restrict__ Bt,
                                                const float* __restrict__ bias,
                                                void* __restrict__ Cout,
                                                int N, int K) {
  __shared__ __align__(16) _Float16 As[2][64 * 32];   // [m][k]
  __shared__ __align__(16) _Float16 Bs[2][64 * 32];   // [n][k]
  int bm = blockIdx.y * 64, bn = blockIdx.x * 64;
  int wid = threadIdx.x >> 5;
  int wm = (wid >> 1) * 32, wn = (wid & 1) * 32;
  v8f acc[2][2] = {};
  const _Float16* Ag = A + (size_t)bm * K;
  const _Float16* Bg = Bt + (size_t)bn * K;
  int nk = K / 32;
  if (wid == 0) {
    tdm_load_2d((unsigned)(uintptr_t)&As[0][0], Ag, 64, 32, (unsigned)K);
    tdm_load_2d((unsigned)(uintptr_t)&Bs[0][0], Bg, 64, 32, (unsigned)K);
  }
  for (int i = 0; i < nk; ++i) {
    if (wid == 0) {
      if (i + 1 < nk) {
        tdm_load_2d((unsigned)(uintptr_t)&As[(i + 1) & 1][0], Ag + (i + 1) * 32,
                    64, 32, (unsigned)K);
        tdm_load_2d((unsigned)(uintptr_t)&Bs[(i + 1) & 1][0], Bg + (i + 1) * 32,
                    64, 32, (unsigned)K);
        __builtin_amdgcn_s_wait_tensorcnt(2);
      } else {
        __builtin_amdgcn_s_wait_tensorcnt(0);
      }
    }
    __syncthreads();
    const _Float16* Ab = &As[i & 1][0];
    const _Float16* Bb = &Bs[i & 1][0];
    v16h a0 = load_frag_NK(Ab, 32, wm, 0);
    v16h a1 = load_frag_NK(Ab, 32, wm + 16, 0);
    v16h b0 = load_frag_NK(Bb, 32, wn, 0);
    v16h b1 = load_frag_NK(Bb, 32, wn + 16, 0);
    acc[0][0] = wmma16(a0, b0, acc[0][0]);
    acc[0][1] = wmma16(a0, b1, acc[0][1]);
    acc[1][0] = wmma16(a1, b0, acc[1][0]);
    acc[1][1] = wmma16(a1, b1, acc[1][1]);
    __syncthreads();
  }
  int lane = threadIdx.x & 31; int cn = lane & 15; int hh = lane >> 4;
#pragma unroll
  for (int j = 0; j < 2; ++j) {
    int n = bn + wn + 16 * j + cn;
    float bv = BIAS ? bias[n] : 0.f;
#pragma unroll
    for (int i = 0; i < 2; ++i)
#pragma unroll
      for (int r = 0; r < 8; ++r) {
        int m = bm + wm + 16 * i + r + 8 * hh;
        float v = acc[i][j][r] + bv;
        if (RELU) v = fmaxf(v, 0.f);
        if (HALF_OUT) ((_Float16*)Cout)[(size_t)m * N + n] = (_Float16)v;
        else          ((float*)Cout)[(size_t)m * N + n] = v;
      }
  }
}

// ---------------------------------------------------------------------------
// RoPE in place on q,k halves of the f16 qkv buffer.
// ---------------------------------------------------------------------------
__global__ void rope_kernel(_Float16* __restrict__ qkv) {
  int idx = blockIdx.x;
  int n = idx >> 4;         // token row
  int h = idx & 15;         // head
  int i = threadIdx.x;      // 0..31 rotation pair
  int t = n % T_;
  float div = expf(-logf(10000.f) * (2.f * (float)i) / (float)HS_);
  float ang = (float)t * div;
  float c = cosf(ang), s = sinf(ang);
  size_t base = (size_t)n * (3 * D_) + (size_t)h * HS_;
  float q1 = (float)qkv[base + i], q2 = (float)qkv[base + 32 + i];
  qkv[base + i]      = (_Float16)(q1 * c - q2 * s);
  qkv[base + 32 + i] = (_Float16)(q2 * c + q1 * s);
  base += D_;
  float k1 = (float)qkv[base + i], k2 = (float)qkv[base + 32 + i];
  qkv[base + i]      = (_Float16)(k1 * c - k2 * s);
  qkv[base + 32 + i] = (_Float16)(k2 * c + k1 * s);
}

// ---------------------------------------------------------------------------
// Flash attention, WMMA. One wave per (b, h, 16-row q tile). Q/K tiles are
// TDM-loaded (K tile DMA overlaps the manual V-transpose copy); the 1/sqrt(HS)
// scale is folded into the logits so Q can be DMA'd raw.
// ---------------------------------------------------------------------------
__global__ __launch_bounds__(32) void attn_kernel(const _Float16* __restrict__ qkv,
                                                  _Float16* __restrict__ o) {
  int qt = blockIdx.x, h = blockIdx.y, b = blockIdx.z;
  __shared__ __align__(16) _Float16 qs[16 * 64];    // [qrow][d]
  __shared__ __align__(16) _Float16 ks[32 * 64];    // [key][d]
  __shared__ __align__(16) _Float16 vsT[64 * 32];   // [d][key]
  __shared__ __align__(16) _Float16 ps[16 * 32];    // [qrow][key]
  int lane = threadIdx.x;
  int cn = lane & 15, hh = lane >> 4;

  const _Float16* qbase = qkv + (size_t)(b * T_ + qt * 16) * (3 * D_) + (size_t)h * HS_;
  tdm_load_2d((unsigned)(uintptr_t)qs, qbase, 16, 64, 3 * D_);
  __builtin_amdgcn_s_wait_tensorcnt(0);
  __syncthreads();

  v8f oacc[4] = {};
  float rmax[8], rsum[8];
#pragma unroll
  for (int r = 0; r < 8; ++r) { rmax[r] = -1e30f; rsum[r] = 0.f; }

  int nkt = (qt * 16 + 16 + 31) / 32;
  for (int kt = 0; kt < nkt; ++kt) {
    const _Float16* kvb = qkv + (size_t)(b * T_ + kt * 32) * (3 * D_) + (size_t)h * HS_;
    tdm_load_2d((unsigned)(uintptr_t)ks, kvb + D_, 32, 64, 3 * D_);   // async K tile
    for (int i = lane; i < 32 * 64; i += 32) {                         // manual V^T
      int r = i >> 6, d = i & 63;
      vsT[d * 32 + r] = kvb[(size_t)r * (3 * D_) + 2 * D_ + d];
      if (kt + 1 < nkt)
        __builtin_prefetch(&kvb[(size_t)(r + 32) * (3 * D_) + 2 * D_ + d], 0, 0);
    }
    __builtin_amdgcn_s_wait_tensorcnt(0);
    __syncthreads();

    v16h qa0 = load_frag_NK(qs, 64, 0, 0);
    v16h qa1 = load_frag_NK(qs, 64, 0, 32);
    v8f sc[2] = {};
#pragma unroll
    for (int g = 0; g < 2; ++g) {
      v16h kb0 = load_frag_NK(ks, 64, 16 * g, 0);
      v16h kb1 = load_frag_NK(ks, 64, 16 * g, 32);
      sc[g] = wmma16(qa0, kb0, sc[g]);
      sc[g] = wmma16(qa1, kb1, sc[g]);
    }

    float tl[2][8];
#pragma unroll
    for (int g = 0; g < 2; ++g)
#pragma unroll
      for (int r = 0; r < 8; ++r) {
        int tq = qt * 16 + r + 8 * hh;
        int tk = kt * 32 + 16 * g + cn;
        float v = sc[g][r] * 0.125f;              // 1/sqrt(HS)
        tl[g][r] = (tk > tq) ? -1e30f : v;
      }

#pragma unroll
    for (int r = 0; r < 8; ++r) {
      float m2 = fmaxf(tl[0][r], tl[1][r]);
      for (int off = 1; off < 16; off <<= 1) m2 = fmaxf(m2, __shfl_xor(m2, off, 16));
      float nm = fmaxf(rmax[r], m2);
      float scale = expf(rmax[r] - nm);
      float psum = 0.f;
#pragma unroll
      for (int g = 0; g < 2; ++g) {
        float p = expf(tl[g][r] - nm);
        tl[g][r] = p;
        psum += p;
      }
      for (int off = 1; off < 16; off <<= 1) psum += __shfl_xor(psum, off, 16);
      rsum[r] = rsum[r] * scale + psum;
      rmax[r] = nm;
#pragma unroll
      for (int g2 = 0; g2 < 4; ++g2) oacc[g2][r] *= scale;
#pragma unroll
      for (int g = 0; g < 2; ++g)
        ps[(r + 8 * hh) * 32 + 16 * g + cn] = (_Float16)tl[g][r];
    }
    __syncthreads();

    v16h pa = load_frag_NK(ps, 32, 0, 0);
#pragma unroll
    for (int g2 = 0; g2 < 4; ++g2) {
      v16h vb = load_frag_NK(vsT, 32, 16 * g2, 0);
      oacc[g2] = wmma16(pa, vb, oacc[g2]);
    }
    __syncthreads();
  }

#pragma unroll
  for (int g2 = 0; g2 < 4; ++g2)
#pragma unroll
    for (int r = 0; r < 8; ++r) {
      int t = qt * 16 + r + 8 * hh;
      int d = 16 * g2 + cn;
      o[(size_t)(b * T_ + t) * D_ + (size_t)h * HS_ + d] = (_Float16)(oacc[g2][r] / rsum[r]);
    }
}

// ---------------------------------------------------------------------------
// x += y
// ---------------------------------------------------------------------------
__global__ void add_kernel(float* __restrict__ x, const float* __restrict__ y, int n) {
  for (int i = blockIdx.x * blockDim.x + threadIdx.x; i < n; i += gridDim.x * blockDim.x)
    x[i] += y[i];
}

// ---------------------------------------------------------------------------
// Router: logits = xn @ w_r + b_r, deterministic top-2 (eps = 0), gates.
// ---------------------------------------------------------------------------
__global__ void route_kernel(const float* __restrict__ xn, const float* __restrict__ w_r,
                             const float* __restrict__ b_r, int* __restrict__ top0,
                             int* __restrict__ top1, float* __restrict__ g0,
                             float* __restrict__ g1) {
  int n = blockIdx.x;
  int e = threadIdx.x >> 5, lane = threadIdx.x & 31;
  float acc = 0.f;
  for (int d = lane; d < D_; d += 32) acc += xn[(size_t)n * D_ + d] * w_r[(size_t)d * E_ + e];
  for (int off = 1; off < 32; off <<= 1) acc += __shfl_xor(acc, off, 32);
  __shared__ float lg[E_];
  if (lane == 0) lg[e] = acc + b_r[e];
  __syncthreads();
  if (threadIdx.x == 0) {
    float v0 = -1e30f, v1 = -1e30f; int i0 = 0, i1 = 0;
    for (int k = 0; k < E_; ++k) {
      float v = lg[k];
      if (v > v0) { v1 = v0; i1 = i0; v0 = v; i0 = k; }
      else if (v > v1) { v1 = v; i1 = k; }
    }
    float s = 1.f + expf(v1 - v0);
    top0[n] = i0; top1[n] = i1;
    g0[n] = 1.f / s;
    g1[n] = expf(v1 - v0) / s;
  }
}

// ---------------------------------------------------------------------------
// Capacity selection: per expert, first CAP tokens (in index order).
// ---------------------------------------------------------------------------
__global__ void select_kernel(const int* __restrict__ top0, const int* __restrict__ top1,
                              const float* __restrict__ g0, const float* __restrict__ g1,
                              int* __restrict__ sel, float* __restrict__ gate) {
  int e = blockIdx.x;
  if (threadIdx.x != 0) return;
  int cnt = 0;
  for (int n = 0; n < NTOK_ && cnt < CAP_; ++n) {
    if (top0[n] == e)      { sel[e * CAP_ + cnt] = n; gate[e * CAP_ + cnt] = g0[n]; ++cnt; }
    else if (top1[n] == e) { sel[e * CAP_ + cnt] = n; gate[e * CAP_ + cnt] = g1[n]; ++cnt; }
  }
  for (; cnt < CAP_; ++cnt) { sel[e * CAP_ + cnt] = -1; gate[e * CAP_ + cnt] = 0.f; }
}

__global__ void gather_kernel(const _Float16* __restrict__ xnh, const int* __restrict__ sel_e,
                              _Float16* __restrict__ xt) {
  int c = blockIdx.x;
  int row = sel_e[c];
  for (int d = threadIdx.x; d < D_; d += 256)
    xt[(size_t)c * D_ + d] = (row >= 0) ? xnh[(size_t)row * D_ + d] : (_Float16)0.f;
}

__global__ void scatter_kernel(const float* __restrict__ eo, const float* __restrict__ gate_e,
                               const int* __restrict__ sel_e, float* __restrict__ x) {
  int c = blockIdx.x;
  int row = sel_e[c];
  if (row < 0) return;
  float g = gate_e[c];
  for (int d = threadIdx.x; d < D_; d += 256)
    x[(size_t)row * D_ + d] += eo[(size_t)c * D_ + d] * g;
}

// ---------------------------------------------------------------------------
// Fused vocab logits + online log-softmax + NLL (WMMA, f16, LDS-resident x).
// Block = 64 threads (2 waves, 32 token rows). The full 32x1024 x-tile stays
// in LDS (64 KB, TDM-loaded once); each vocab step TDM-loads one 16x1024 emb
// tile (32 KB) and runs 32 barrier-free WMMAs per wave against it.
// ---------------------------------------------------------------------------
__global__ __launch_bounds__(64) void nll_kernel(const _Float16* __restrict__ xf,
                                                 const _Float16* __restrict__ emb,
                                                 const int* __restrict__ ids,
                                                 float* __restrict__ accum) {
  __shared__ __align__(16) _Float16 xs[32 * 1024];   // 64 KB
  __shared__ __align__(16) _Float16 es[16 * 1024];   // 32 KB
  int wid = threadIdx.x >> 5, lane = threadIdx.x & 31;
  int cn = lane & 15, hh = lane >> 4;
  int rbase = blockIdx.x * 32;

  if (wid == 0) {
    tdm_load_2d((unsigned)(uintptr_t)xs, xf + (size_t)rbase * D_, 32, 1024, D_);
    __builtin_amdgcn_s_wait_tensorcnt(0);
  }
  __syncthreads();

  float rmax[8], rsum[8], tlog[8];
#pragma unroll
  for (int r = 0; r < 8; ++r) { rmax[r] = -1e30f; rsum[r] = 0.f; tlog[r] = 0.f; }

  int tgt_r[8];
#pragma unroll
  for (int r = 0; r < 8; ++r) {
    int row = rbase + wid * 16 + r + 8 * hh;
    int t = row % T_;
    tgt_r[r] = (t < T_ - 1) ? ids[row + 1] : -1;
  }

  for (int vt = 0; vt < V_ / 16; ++vt) {
    if (wid == 0) {
      tdm_load_2d((unsigned)(uintptr_t)es, emb + (size_t)(vt * 16) * D_, 16, 1024, D_);
      __builtin_amdgcn_s_wait_tensorcnt(0);
    }
    __syncthreads();
    v8f acc = {};
#pragma unroll 4
    for (int kc = 0; kc < D_ / 32; ++kc) {
      v16h a = load_frag_NK(xs, 1024, wid * 16, kc * 32);
      v16h b = load_frag_NK(es, 1024, 0, kc * 32);
      acc = wmma16(a, b, acc);
    }
    int col = vt * 16 + cn;
#pragma unroll
    for (int r = 0; r < 8; ++r) {
      float v = acc[r];
      float tv = (col == tgt_r[r]) ? v : 0.f;
      for (int off = 1; off < 16; off <<= 1) tv += __shfl_xor(tv, off, 16);
      tlog[r] += tv;
      float m2 = v;
      for (int off = 1; off < 16; off <<= 1) m2 = fmaxf(m2, __shfl_xor(m2, off, 16));
      float nm = fmaxf(rmax[r], m2);
      float scl = expf(rmax[r] - nm);
      float p = expf(v - nm);
      for (int off = 1; off < 16; off <<= 1) p += __shfl_xor(p, off, 16);
      rsum[r] = rsum[r] * scl + p;
      rmax[r] = nm;
    }
    __syncthreads();   // before next emb tile overwrites es
  }

  if (cn == 0) {
#pragma unroll
    for (int r = 0; r < 8; ++r) {
      int tgt = tgt_r[r];
      if (tgt >= 0 && tgt != PAD_) {
        float logp = tlog[r] - rmax[r] - logf(rsum[r]);
        atomicAdd(&accum[0], -logp);
        atomicAdd(&accum[1], 1.f);
      }
    }
  }
}

__global__ void zero_accum_kernel(float* a) {
  if (threadIdx.x < 2) a[threadIdx.x] = 0.f;
}

__global__ void finalize_kernel(const float* __restrict__ accum, float* __restrict__ out) {
  out[0] = accum[0] / fmaxf(accum[1], 1.f);
}

// ---------------------------------------------------------------------------
// Host launcher
// ---------------------------------------------------------------------------
extern "C" void kernel_launch(void* const* d_in, const int* in_sizes, int n_in,
                              void* d_out, int out_size, void* d_ws, size_t ws_size,
                              hipStream_t stream) {
  (void)in_sizes; (void)n_in; (void)out_size; (void)ws_size;
  const int*   ids     = (const int*)d_in[0];
  const float* tok_emb = (const float*)d_in[1];
  const float* pos_emb = (const float*)d_in[2];
  const float* ln1_w   = (const float*)d_in[3];
  const float* ln1_b   = (const float*)d_in[4];
  const float* ln2_w   = (const float*)d_in[5];
  const float* ln2_b   = (const float*)d_in[6];
  const float* w_qkv   = (const float*)d_in[7];
  const float* w_out   = (const float*)d_in[8];
  const float* w_route = (const float*)d_in[9];
  const float* b_route = (const float*)d_in[10];
  // d_in[11], d_in[12]: w_noise/b_noise unused (deterministic eps = 0)
  const float* w_e1    = (const float*)d_in[13];
  const float* b_e1    = (const float*)d_in[14];
  const float* w_e2    = (const float*)d_in[15];
  const float* b_e2    = (const float*)d_in[16];
  const float* lnf_w   = (const float*)d_in[17];
  const float* lnf_b   = (const float*)d_in[18];

  char* base = (char*)d_ws;
  size_t off = 0;
  auto alloc = [&](size_t bytes) -> void* {
    void* r = base + off;
    off = (off + bytes + 255) & ~(size_t)255;
    return r;
  };
  float*     x     = (float*)alloc((size_t)NTOK_ * D_ * 4);
  float*     xn    = (float*)alloc((size_t)NTOK_ * D_ * 4);
  _Float16*  xnh   = (_Float16*)alloc((size_t)NTOK_ * D_ * 2);
  _Float16*  qkv   = (_Float16*)alloc((size_t)NTOK_ * 3 * D_ * 2);
  _Float16*  oh    = (_Float16*)alloc((size_t)NTOK_ * D_ * 2);
  float*     proj  = (float*)alloc((size_t)NTOK_ * D_ * 4);
  _Float16*  xt    = (_Float16*)alloc((size_t)CAP_ * D_ * 2);
  _Float16*  hbuf  = (_Float16*)alloc((size_t)CAP_ * FF_ * 2);
  float*     eo    = (float*)alloc((size_t)CAP_ * D_ * 4);
  _Float16*  embh  = (_Float16*)alloc((size_t)V_ * D_ * 2);
  _Float16*  wt    = (_Float16*)alloc((size_t)FF_ * D_ * 2);   // max [N][K] = 4096x1024
  int*       top0  = (int*)alloc((size_t)NTOK_ * 4);
  int*       top1  = (int*)alloc((size_t)NTOK_ * 4);
  float*     g0    = (float*)alloc((size_t)NTOK_ * 4);
  float*     g1    = (float*)alloc((size_t)NTOK_ * 4);
  int*       sel   = (int*)alloc((size_t)E_ * CAP_ * 4);
  float*     gate  = (float*)alloc((size_t)E_ * CAP_ * 4);
  float*     accum = (float*)alloc(2 * 4);

  embed_kernel<<<NTOK_, 256, 0, stream>>>(ids, tok_emb, pos_emb, x);
  convert_kernel<<<4096, 256, 0, stream>>>(tok_emb, embh, V_ * D_);

  for (int l = 0; l < L_; ++l) {
    // Attention block
    ln_kernel<<<NTOK_, 256, 0, stream>>>(x, ln1_w + (size_t)l * D_, ln1_b + (size_t)l * D_,
                                         xn, xnh);
    transpose_conv_kernel<<<dim3(3 * D_ / 32, D_ / 32), 256, 0, stream>>>(
        w_qkv + (size_t)l * D_ * 3 * D_, wt, D_, 3 * D_);
    gemm_tdm<false, false, true><<<dim3(3 * D_ / 64, NTOK_ / 64), 128, 0, stream>>>(
        xnh, wt, nullptr, qkv, 3 * D_, D_);
    rope_kernel<<<NTOK_ * H_, 32, 0, stream>>>(qkv);
    attn_kernel<<<dim3(T_ / 16, H_, B_), 32, 0, stream>>>(qkv, oh);
    transpose_conv_kernel<<<dim3(D_ / 32, D_ / 32), 256, 0, stream>>>(
        w_out + (size_t)l * D_ * D_, wt, D_, D_);
    gemm_tdm<false, false, false><<<dim3(D_ / 64, NTOK_ / 64), 128, 0, stream>>>(
        oh, wt, nullptr, proj, D_, D_);
    add_kernel<<<1024, 256, 0, stream>>>(x, proj, NTOK_ * D_);

    // MoE block
    ln_kernel<<<NTOK_, 256, 0, stream>>>(x, ln2_w + (size_t)l * D_, ln2_b + (size_t)l * D_,
                                         xn, xnh);
    route_kernel<<<NTOK_, 256, 0, stream>>>(
        xn, w_route + (size_t)l * D_ * E_, b_route + (size_t)l * E_, top0, top1, g0, g1);
    select_kernel<<<E_, 32, 0, stream>>>(top0, top1, g0, g1, sel, gate);
    for (int e = 0; e < E_; ++e) {
      gather_kernel<<<CAP_, 256, 0, stream>>>(xnh, sel + (size_t)e * CAP_, xt);
      transpose_conv_kernel<<<dim3(FF_ / 32, D_ / 32), 256, 0, stream>>>(
          w_e1 + ((size_t)l * E_ + e) * D_ * FF_, wt, D_, FF_);
      gemm_tdm<true, true, true><<<dim3(FF_ / 64, CAP_ / 64), 128, 0, stream>>>(
          xt, wt, b_e1 + ((size_t)l * E_ + e) * FF_, hbuf, FF_, D_);
      transpose_conv_kernel<<<dim3(D_ / 32, FF_ / 32), 256, 0, stream>>>(
          w_e2 + ((size_t)l * E_ + e) * FF_ * D_, wt, FF_, D_);
      gemm_tdm<true, false, false><<<dim3(D_ / 64, CAP_ / 64), 128, 0, stream>>>(
          hbuf, wt, b_e2 + ((size_t)l * E_ + e) * D_, eo, D_, FF_);
      scatter_kernel<<<CAP_, 256, 0, stream>>>(eo, gate + (size_t)e * CAP_,
                                               sel + (size_t)e * CAP_, x);
    }
  }

  // Final LN + fused vocab NLL
  ln_kernel<<<NTOK_, 256, 0, stream>>>(x, lnf_w, lnf_b, xn, xnh);
  zero_accum_kernel<<<1, 32, 0, stream>>>(accum);
  nll_kernel<<<NTOK_ / 32, 64, 0, stream>>>(xnh, embh, ids, accum);
  finalize_kernel<<<1, 1, 0, stream>>>(accum, (float*)d_out);
}
